// FFTFormer_57836029608188
// MI455X (gfx1250) — compile-verified, matrix operand order checked by reading, not run on hardware
//
#include <hip/hip_runtime.h>
#include <math.h>

// ---------------------------------------------------------------------------
// FFTFormer forward for MI455X (gfx1250, wave32).
//  - 1x1 convs  -> v_wmma_f32_16x16x32_f16 GEMM (f16 in, f32 accumulate).
//    Activations transposed+cast to f16 [N][K] once per GEMM; each wave
//    computes a 16x64 output block (A fragment reused across 4 WMMAs,
//    all fragment loads are contiguous global_load_b128).
//  - 8x8 patch FFTs -> 8-thread-per-patch DFT with LDS transposes (ds ops)
//  - everything else -> bandwidth-bound elementwise / small convs (L2-resident)
// ---------------------------------------------------------------------------

typedef __attribute__((ext_vector_type(16))) _Float16 v16h;
typedef __attribute__((ext_vector_type(8)))  _Float16 v8h;
typedef __attribute__((ext_vector_type(8)))  float    v8f;

static inline int cdiv_ll(long long a, long long b) { return (int)((a + b - 1) / b); }

__device__ __forceinline__ v16h cat16(v8h lo, v8h hi)
{
  return __builtin_shufflevector(lo, hi, 0, 1, 2, 3, 4, 5, 6, 7,
                                         8, 9, 10, 11, 12, 13, 14, 15);
}

// ========================= f16 cast / transpose ============================
__global__ __launch_bounds__(256) void k_cast_f16(
    const float* __restrict__ x, _Float16* __restrict__ y, long long n)
{
  const long long i = (long long)blockIdx.x * blockDim.x + threadIdx.x;
  if (i < n) y[i] = (_Float16)x[i];
}

// X f32 [K][N] -> XT f16 [N][K], 32x32 LDS tiles. N % 32 == 0; K % 16 == 0.
__global__ __launch_bounds__(256) void k_transpose_cast(
    const float* __restrict__ X, _Float16* __restrict__ XT, int K, int N)
{
  __shared__ float t[32][33];
  const int tx = threadIdx.x & 31;
  const int ty = threadIdx.x >> 5;            // 0..7
  const int ktiles = (K + 31) >> 5;
  const int bk = blockIdx.x % ktiles;
  const int bn = blockIdx.x / ktiles;
  const int k0 = bk * 32, n0 = bn * 32;
#pragma unroll
  for (int j = 0; j < 4; ++j) {
    const int k = k0 + ty + j * 8;
    t[ty + j * 8][tx] = (k < K) ? X[(size_t)k * N + (n0 + tx)] : 0.f;
  }
  __syncthreads();
#pragma unroll
  for (int j = 0; j < 4; ++j) {
    const int n = n0 + ty + j * 8;
    const int k = k0 + tx;
    if (k < K) XT[(size_t)n * K + k] = (_Float16)t[tx][ty + j * 8];
  }
}

// ============================ WMMA GEMM ====================================
// Y[M,N] = Wh[M,K] * XT[N,K]^T (+bias[M]) (+add[M,N]).
// M%16==0, K%16==0, N%64==0. One wave per 16x64 output block: the A (weight)
// fragment is reused across 4 WMMAs per K-step.
__global__ __launch_bounds__(256) void k_gemm_wmma(
    const _Float16* __restrict__ Wh, const _Float16* __restrict__ XT,
    const float* __restrict__ bias, const float* __restrict__ add,
    float* __restrict__ Y, int M, int K, int N, long long nWaves)
{
  const long long wave = (long long)blockIdx.x * 8 + (threadIdx.x >> 5);
  if (wave >= nWaves) return;                 // whole-wave exit: EXEC stays full
  const int lane = threadIdx.x & 31;
  const int nt64 = N >> 6;
  const int mt = (int)(wave / nt64);
  const int nb = (int)(wave % nt64);
  const int row  = lane & 15;                 // A row / B column within tile
  const int half = (lane >> 4) << 4;          // K sub-block: 0 or 16

  const _Float16* __restrict__ ap = Wh + (size_t)(mt * 16 + row) * K + half;
  const _Float16* __restrict__ bp = XT + (size_t)(nb * 64 + row) * K + half;
  const size_t bstr = (size_t)16 * K;         // 16 columns of XT

  v8f acc[4] = {{}, {}, {}, {}};
  const int K32 = K & ~31;
  for (int k0 = 0; k0 < K32; k0 += 32) {
    const v16h a = cat16(*(const v8h*)(ap + k0), *(const v8h*)(ap + k0 + 8));
    __builtin_prefetch(bp + k0 + 128, 0, 1);            // global_prefetch_b8
#pragma unroll
    for (int j = 0; j < 4; ++j) {
      const v16h b = cat16(*(const v8h*)(bp + j * bstr + k0),
                           *(const v8h*)(bp + j * bstr + k0 + 8));
      acc[j] = __builtin_amdgcn_wmma_f32_16x16x32_f16(
          false, a, false, b, (short)0, acc[j], false, false);
    }
  }
  if (K & 16) {                 // 16-wide K tail: upper-half lanes supply zeros
    v16h a = {}, b[4] = {{}, {}, {}, {}};
    if (half == 0) {
      a = cat16(*(const v8h*)(ap + K32), *(const v8h*)(ap + K32 + 8));
#pragma unroll
      for (int j = 0; j < 4; ++j)
        b[j] = cat16(*(const v8h*)(bp + j * bstr + K32),
                     *(const v8h*)(bp + j * bstr + K32 + 8));
    }
#pragma unroll
    for (int j = 0; j < 4; ++j)
      acc[j] = __builtin_amdgcn_wmma_f32_16x16x32_f16(
          false, a, false, b[j], (short)0, acc[j], false, false);
  }
  // D layout: VGPR r -> M = r + 8*(lane>=16), N = lane&15   (ISA 7.12.2)
  const int mbase = mt * 16 + ((lane >> 4) << 3);
#pragma unroll
  for (int j = 0; j < 4; ++j) {
    const int cc = nb * 64 + j * 16 + (lane & 15);
#pragma unroll
    for (int r = 0; r < 8; ++r) {
      const int    mm  = mbase + r;
      const size_t idx = (size_t)mm * N + cc;
      float v = acc[j][r];
      if (bias) v += bias[mm];
      if (add)  v += add[idx];
      Y[idx] = v;
    }
  }
}

// ================================ DFT-8 ====================================
__device__ __constant__ float C8[8] = { 1.f,  0.70710678f,  0.f, -0.70710678f,
                                       -1.f, -0.70710678f,  0.f,  0.70710678f};
__device__ __constant__ float S8[8] = { 0.f,  0.70710678f,  1.f,  0.70710678f,
                                        0.f, -0.70710678f, -1.f, -0.70710678f};

__device__ __forceinline__ void dft8(const float* xr, const float* xi,
                                     float* yr, float* yi, float sgn, float scl)
{
#pragma unroll
  for (int k = 0; k < 8; ++k) {
    float sr = 0.f, si = 0.f;
#pragma unroll
    for (int t = 0; t < 8; ++t) {
      const int   a = (k * t) & 7;
      const float c = C8[a];
      const float s = sgn * S8[a];
      sr += xr[t] * c - xi[t] * s;
      si += xr[t] * s + xi[t] * c;
    }
    yr[k] = sr * scl; yi[k] = si * scl;
  }
}

// ============== dffn spectral modulation: x = irfft2(rfft2(x)*filt) ========
// 8 threads per 8x8 patch, 32 patches per 256-thread block. In-place.
__global__ __launch_bounds__(256) void k_fft_mod(
    float* __restrict__ x, const float* __restrict__ filt, int C, int H, int W)
{
  __shared__ float lr[32][8][9];
  __shared__ float li[32][8][9];
  const int tid = threadIdx.x, p = tid >> 3, r = tid & 7;
  const int pw = W >> 3;
  const long long ppc = (long long)(H >> 3) * pw;
  const long long gp  = (long long)blockIdx.x * 32 + p;
  const int c  = (int)(gp / ppc);
  const int lp = (int)(gp % ppc);
  const int py = lp / pw, px = lp % pw;
  float* __restrict__ base = x + ((size_t)c * H + (size_t)py * 8) * W + (size_t)px * 8;

  float ar[8], ai[8], br[8], bi[8];
#pragma unroll
  for (int t = 0; t < 8; ++t) { ar[t] = base[(size_t)r * W + t]; ai[t] = 0.f; }
  dft8(ar, ai, br, bi, -1.f, 1.f);
#pragma unroll
  for (int v = 0; v < 8; ++v) { lr[p][v][r] = br[v]; li[p][v][r] = bi[v]; }
  __syncthreads();
#pragma unroll
  for (int y = 0; y < 8; ++y) { ar[y] = lr[p][r][y]; ai[y] = li[p][r][y]; }
  dft8(ar, ai, br, bi, -1.f, 1.f);
  {
    const int v  = r;
    const int vp = (v <= 4) ? v : 8 - v;
#pragma unroll
    for (int u = 0; u < 8; ++u) {
      const int   up = (v <= 4) ? u : ((8 - u) & 7);
      const float sc = filt[((size_t)c * 8 + up) * 5 + vp];
      br[u] *= sc; bi[u] *= sc;
    }
  }
  dft8(br, bi, ar, ai, +1.f, 0.125f);
  __syncthreads();
#pragma unroll
  for (int y = 0; y < 8; ++y) { lr[p][y][r] = ar[y]; li[p][y][r] = ai[y]; }
  __syncthreads();
#pragma unroll
  for (int v = 0; v < 8; ++v) { ar[v] = lr[p][r][v]; ai[v] = li[p][r][v]; }
  dft8(ar, ai, br, bi, +1.f, 0.125f);
#pragma unroll
  for (int t = 0; t < 8; ++t) base[(size_t)r * W + t] = br[t];
}

// ============== FSAS spectral correlation: o = irfft2(rfft2(q)*rfft2(k)) ===
__global__ __launch_bounds__(256) void k_fft_qk(
    const float* __restrict__ q, const float* __restrict__ kin,
    float* __restrict__ o, int C, int H, int W)
{
  __shared__ float lr[32][8][9];
  __shared__ float li[32][8][9];
  const int tid = threadIdx.x, p = tid >> 3, r = tid & 7;
  const int pw = W >> 3;
  const long long ppc = (long long)(H >> 3) * pw;
  const long long gp  = (long long)blockIdx.x * 32 + p;
  const int c  = (int)(gp / ppc);
  const int lp = (int)(gp % ppc);
  const int py = lp / pw, px = lp % pw;
  const size_t off = ((size_t)c * H + (size_t)py * 8) * W + (size_t)px * 8;

  float ar[8], ai[8], br[8], bi[8], fr[8], fi[8];
#pragma unroll
  for (int t = 0; t < 8; ++t) { ar[t] = q[off + (size_t)r * W + t]; ai[t] = 0.f; }
  dft8(ar, ai, br, bi, -1.f, 1.f);
#pragma unroll
  for (int v = 0; v < 8; ++v) { lr[p][v][r] = br[v]; li[p][v][r] = bi[v]; }
  __syncthreads();
#pragma unroll
  for (int y = 0; y < 8; ++y) { ar[y] = lr[p][r][y]; ai[y] = li[p][r][y]; }
  dft8(ar, ai, fr, fi, -1.f, 1.f);
  __syncthreads();
#pragma unroll
  for (int t = 0; t < 8; ++t) { ar[t] = kin[off + (size_t)r * W + t]; ai[t] = 0.f; }
  dft8(ar, ai, br, bi, -1.f, 1.f);
#pragma unroll
  for (int v = 0; v < 8; ++v) { lr[p][v][r] = br[v]; li[p][v][r] = bi[v]; }
  __syncthreads();
#pragma unroll
  for (int y = 0; y < 8; ++y) { ar[y] = lr[p][r][y]; ai[y] = li[p][r][y]; }
  dft8(ar, ai, br, bi, -1.f, 1.f);
#pragma unroll
  for (int u = 0; u < 8; ++u) {
    const float prr = fr[u] * br[u] - fi[u] * bi[u];
    const float pii = fr[u] * bi[u] + fi[u] * br[u];
    ar[u] = prr; ai[u] = pii;
  }
  dft8(ar, ai, br, bi, +1.f, 0.125f);
  __syncthreads();
#pragma unroll
  for (int y = 0; y < 8; ++y) { lr[p][y][r] = br[y]; li[p][y][r] = bi[y]; }
  __syncthreads();
#pragma unroll
  for (int v = 0; v < 8; ++v) { ar[v] = lr[p][r][v]; ai[v] = li[p][r][v]; }
  dft8(ar, ai, br, bi, +1.f, 0.125f);
#pragma unroll
  for (int t = 0; t < 8; ++t) o[off + (size_t)r * W + t] = br[t];
}

// =========================== simple kernels ================================
// LayerNorm over contiguous chunks of C floats (faithful to the raw reshape).
__global__ __launch_bounds__(256) void k_layernorm(
    const float* __restrict__ x, const float* __restrict__ w,
    const float* __restrict__ b, float* __restrict__ y, int C, long long G)
{
  const long long g = (long long)blockIdx.x * blockDim.x + threadIdx.x;
  if (g >= G) return;
  const float* p = x + (size_t)g * C;
  float s = 0.f, s2 = 0.f;
  for (int c = 0; c < C; ++c) { const float v = p[c]; s += v; s2 += v * v; }
  const float mu   = s / C;
  const float var  = s2 / C - mu * mu;
  const float rstd = rsqrtf(var + 1e-5f);
  float* q = y + (size_t)g * C;
  for (int c = 0; c < C; ++c) q[c] = (p[c] - mu) * rstd * w[c] + b[c];
}

__global__ __launch_bounds__(256) void k_dwconv3(
    const float* __restrict__ x, const float* __restrict__ w,
    float* __restrict__ y, int C, int H, int W)
{
  const long long i = (long long)blockIdx.x * blockDim.x + threadIdx.x;
  const long long tot = (long long)C * H * W;
  if (i >= tot) return;
  const int xw = (int)(i % W);
  const int yh = (int)((i / W) % H);
  const int c  = (int)(i / ((long long)W * H));
  const float* wp = w + (size_t)c * 9;
  const float* xp = x + (size_t)c * H * W;
  float acc = 0.f;
#pragma unroll
  for (int ky = 0; ky < 3; ++ky) {
    const int yy = yh + ky - 1;
    if (yy < 0 || yy >= H) continue;
#pragma unroll
    for (int kx = 0; kx < 3; ++kx) {
      const int xx = xw + kx - 1;
      if (xx < 0 || xx >= W) continue;
      acc = fmaf(wp[ky * 3 + kx], xp[(size_t)yy * W + xx], acc);
    }
  }
  y[i] = acc;
}

__global__ __launch_bounds__(256) void k_conv3x3(
    const float* __restrict__ x, const float* __restrict__ w,
    const float* __restrict__ bias, float* __restrict__ y,
    int Cin, int Cout, int H, int W, const float* __restrict__ resid)
{
  const long long i = (long long)blockIdx.x * blockDim.x + threadIdx.x;
  const long long tot = (long long)Cout * H * W;
  if (i >= tot) return;
  const int xw = (int)(i % W);
  const int yh = (int)((i / W) % H);
  const int co = (int)(i / ((long long)W * H));
  float acc = bias ? bias[co] : 0.f;
  for (int ci = 0; ci < Cin; ++ci) {
    const float* xp = x + (size_t)ci * H * W;
    const float* wp = w + ((size_t)co * Cin + ci) * 9;
#pragma unroll
    for (int ky = 0; ky < 3; ++ky) {
      const int yy = yh + ky - 1;
      if (yy < 0 || yy >= H) continue;
#pragma unroll
      for (int kx = 0; kx < 3; ++kx) {
        const int xx = xw + kx - 1;
        if (xx < 0 || xx >= W) continue;
        acc = fmaf(wp[ky * 3 + kx], xp[(size_t)yy * W + xx], acc);
      }
    }
  }
  if (resid) acc += resid[i];
  y[i] = acc;
}

// bilinear, align_corners=False, no antialias
__global__ __launch_bounds__(256) void k_resize(
    const float* __restrict__ x, float* __restrict__ y,
    int C, int Hi, int Wi, int Ho, int Wo)
{
  const long long i = (long long)blockIdx.x * blockDim.x + threadIdx.x;
  const long long tot = (long long)C * Ho * Wo;
  if (i >= tot) return;
  const int ox = (int)(i % Wo);
  const int oy = (int)((i / Wo) % Ho);
  const int c  = (int)(i / ((long long)Wo * Ho));
  const float fy = ((oy + 0.5f) * Hi) / Ho - 0.5f;
  const float fx = ((ox + 0.5f) * Wi) / Wo - 0.5f;
  const int y0 = (int)floorf(fy), x0 = (int)floorf(fx);
  const float wy = fy - y0, wx = fx - x0;
  const int y0c = min(max(y0, 0), Hi - 1), y1c = min(max(y0 + 1, 0), Hi - 1);
  const int x0c = min(max(x0, 0), Wi - 1), x1c = min(max(x0 + 1, 0), Wi - 1);
  const float* xp = x + (size_t)c * Hi * Wi;
  const float a = xp[(size_t)y0c * Wi + x0c], b = xp[(size_t)y0c * Wi + x1c];
  const float d = xp[(size_t)y1c * Wi + x0c], e = xp[(size_t)y1c * Wi + x1c];
  y[i] = (a * (1.f - wx) + b * wx) * (1.f - wy) + (d * (1.f - wx) + e * wx) * wy;
}

__global__ __launch_bounds__(256) void k_gelu_gate(
    const float* __restrict__ x, float* __restrict__ y, long long n)
{
  const long long i = (long long)blockIdx.x * blockDim.x + threadIdx.x;
  if (i >= n) return;
  const float a = x[i];
  const float g = 0.5f * a * (1.f + erff(a * 0.70710678f));
  y[i] = g * x[i + n];
}

__global__ __launch_bounds__(256) void k_split_add(
    const float* __restrict__ x, float* __restrict__ y, long long n)
{
  const long long i = (long long)blockIdx.x * blockDim.x + threadIdx.x;
  if (i >= n) return;
  y[i] = x[i] + x[i + n];
}

__global__ __launch_bounds__(256) void k_mul_inplace(
    float* __restrict__ o, const float* __restrict__ v, long long n)
{
  const long long i = (long long)blockIdx.x * blockDim.x + threadIdx.x;
  if (i >= n) return;
  o[i] *= v[i];
}

__global__ __launch_bounds__(256) void k_copy(
    const float* __restrict__ a, float* __restrict__ b, long long n)
{
  const long long i = (long long)blockIdx.x * blockDim.x + threadIdx.x;
  if (i >= n) return;
  b[i] = a[i];
}

// ============================ host orchestration ===========================
namespace {

struct P {                               // linear cursor over flattened params
  void* const* in; int i;
  const float* n() { return (const float*)in[i++]; }
};

struct Alloc {                           // bump allocator over d_ws
  char* base; size_t off;
  float* f(size_t nfloats) {
    const size_t o = off;
    off += ((nfloats * sizeof(float)) + 255) & ~(size_t)255;
    return (float*)(base + o);
  }
  _Float16* h(size_t nhalfs) {
    const size_t o = off;
    off += ((nhalfs * sizeof(_Float16)) + 255) & ~(size_t)255;
    return (_Float16*)(base + o);
  }
};

// Y[M,N] = Wm[M,K] * X[K,N] (+bias)(+add): cast W, transpose+cast X, WMMA GEMM
void conv1x1(const float* Wm, const float* X, const float* bias,
             const float* add, float* Y, int M, int K, int N,
             Alloc al, hipStream_t s)
{
  _Float16* Wh = al.h((size_t)M * K);
  k_cast_f16<<<cdiv_ll((long long)M * K, 256), 256, 0, s>>>(Wm, Wh, (long long)M * K);
  _Float16* XT = al.h((size_t)N * K);
  const int ktiles = (K + 31) / 32;
  k_transpose_cast<<<ktiles * (N / 32), 256, 0, s>>>(X, XT, K, N);
  const long long nWaves = (long long)((M + 15) / 16) * (N / 64);
  k_gemm_wmma<<<cdiv_ll(nWaves, 8), 256, 0, s>>>(Wh, XT, bias, add, Y, M, K, N, nWaves);
}

void dffn(P& p, const float* xin, float* xacc, int C, int H, int W,
          Alloc al, hipStream_t s)
{
  const float* pin  = p.n();
  const float* dw   = p.n();
  const float* fft  = p.n();
  const float* pout = p.n();
  const int    Hh = C * 3, C2 = 2 * Hh;
  const long long HW = (long long)H * W;
  float* t2 = al.f((size_t)C2 * HW);
  conv1x1(pin, xin, nullptr, nullptr, t2, C2, C, (int)HW, al, s);
  k_fft_mod<<<(int)(((long long)C2 * (HW / 64)) / 32), 256, 0, s>>>(t2, fft, C2, H, W);
  float* t3 = al.f((size_t)C2 * HW);
  k_dwconv3<<<cdiv_ll((long long)C2 * HW, 256), 256, 0, s>>>(t2, dw, t3, C2, H, W);
  k_gelu_gate<<<cdiv_ll((long long)Hh * HW, 256), 256, 0, s>>>(t3, t2, (long long)Hh * HW);
  conv1x1(pout, t2, nullptr, xacc, xacc, C, Hh, (int)HW, al, s);
}

void fsas(P& p, const float* xin, float* xacc, int C, int H, int W,
          Alloc al, hipStream_t s)
{
  const float* pin  = p.n();
  const float* dw   = p.n();
  const float* pout = p.n();
  const int    C6 = 6 * C, C2 = 2 * C;
  const long long HW = (long long)H * W;
  float* t1 = al.f((size_t)C6 * HW);
  conv1x1(pin, xin, nullptr, nullptr, t1, C6, C, (int)HW, al, s);
  float* t2 = al.f((size_t)C6 * HW);
  k_dwconv3<<<cdiv_ll((long long)C6 * HW, 256), 256, 0, s>>>(t1, dw, t2, C6, H, W);
  const float* q = t2;
  const float* k = t2 + (size_t)C2 * HW;
  const float* v = t2 + (size_t)2 * C2 * HW;
  float* o = t1;  // reuse pin output as correlation buffer
  k_fft_qk<<<(int)(((long long)C2 * (HW / 64)) / 32), 256, 0, s>>>(q, k, o, C2, H, W);
  k_mul_inplace<<<cdiv_ll((long long)C2 * HW, 256), 256, 0, s>>>(o, v, (long long)C2 * HW);
  conv1x1(pout, o, nullptr, xacc, xacc, C, C2, (int)HW, al, s);
}

void tlayer(P& p, float* x, int C, int H, int W, Alloc al, hipStream_t s)
{
  const float* nw = p.n(); const float* nb = p.n();
  const long long HW = (long long)H * W;
  float* t = al.f((size_t)C * HW);
  k_layernorm<<<cdiv_ll(HW, 256), 256, 0, s>>>(x, nw, nb, t, C, HW);
  dffn(p, t, x, C, H, W, al, s);
}

void talayer(P& p, float* x, int C, int H, int W, Alloc al, hipStream_t s)
{
  const long long HW = (long long)H * W;
  float* t = al.f((size_t)C * HW);
  { const float* nw = p.n(); const float* nb = p.n();
    k_layernorm<<<cdiv_ll(HW, 256), 256, 0, s>>>(x, nw, nb, t, C, HW);
    fsas(p, t, x, C, H, W, al, s); }
  { const float* nw = p.n(); const float* nb = p.n();
    k_layernorm<<<cdiv_ll(HW, 256), 256, 0, s>>>(x, nw, nb, t, C, HW);
    dffn(p, t, x, C, H, W, al, s); }
}

// resize then 3x3 conv (encoder down / decoder up)
void scale_conv(P& p, const float* x, float* out, int Cin, int Cout,
                int H, int W, int Ho, int Wo, Alloc al, hipStream_t s)
{
  float* t = al.f((size_t)Cin * Ho * Wo);
  k_resize<<<cdiv_ll((long long)Cin * Ho * Wo, 256), 256, 0, s>>>(x, t, Cin, H, W, Ho, Wo);
  const float* w = p.n(); const float* b = p.n();
  k_conv3x3<<<cdiv_ll((long long)Cout * Ho * Wo, 256), 256, 0, s>>>(
      t, w, b, out, Cin, Cout, Ho, Wo, nullptr);
}

void fuse(P& p, const float* d, const float* e, float* out, int nf,
          int H, int W, Alloc al, hipStream_t s)
{
  const int C2 = 2 * nf;
  const long long HW = (long long)H * W;
  float* cat = al.f((size_t)C2 * HW);
  k_copy<<<cdiv_ll((long long)nf * HW, 256), 256, 0, s>>>(d, cat, (long long)nf * HW);
  k_copy<<<cdiv_ll((long long)nf * HW, 256), 256, 0, s>>>(e, cat + (size_t)nf * HW, (long long)nf * HW);
  const float* w1 = p.n(); const float* b1 = p.n();
  float* t = al.f((size_t)C2 * HW);
  conv1x1(w1, cat, b1, nullptr, t, C2, C2, (int)HW, al, s);
  tlayer(p, t, C2, H, W, al, s);
  const float* w2 = p.n(); const float* b2 = p.n();
  conv1x1(w2, t, b2, nullptr, cat, C2, C2, (int)HW, al, s);   // reuse cat
  k_split_add<<<cdiv_ll((long long)nf * HW, 256), 256, 0, s>>>(cat, out, (long long)nf * HW);
}

} // namespace

extern "C" void kernel_launch(void* const* d_in, const int* in_sizes, int n_in,
                              void* d_out, int out_size, void* d_ws, size_t ws_size,
                              hipStream_t stream)
{
  (void)in_sizes; (void)n_in; (void)out_size; (void)ws_size;
  P p{d_in, 0};
  const float* img = (const float*)d_in[p.i++];
  Alloc al{(char*)d_ws, 0};

  // persistent activations
  float* x0 = al.f((size_t)48  * 256 * 256);   // encoder 0 out (skip)
  float* x1 = al.f((size_t)96  * 128 * 128);   // encoder 1 out (skip)
  float* x2 = al.f((size_t)192 * 64  * 64);    // encoder 2 / middle (in-place)
  float* d1 = al.f((size_t)96  * 128 * 128);
  float* f1 = al.f((size_t)96  * 128 * 128);
  float* d0 = al.f((size_t)48  * 256 * 256);
  float* f0 = al.f((size_t)48  * 256 * 256);

  // ---- embed: 3x3 conv 3 -> 48 ----
  { const float* w = p.n(); const float* b = p.n();
    k_conv3x3<<<cdiv_ll((long long)48 * 65536, 256), 256, 0, stream>>>(
        img, w, b, x0, 3, 48, 256, 256, nullptr); }

  // ---- encoders ----
  for (int l = 0; l < 2; ++l) tlayer(p, x0, 48, 256, 256, al, stream);
  scale_conv(p, x0, x1, 48, 96, 256, 256, 128, 128, al, stream);      // down
  for (int l = 0; l < 2; ++l) tlayer(p, x1, 96, 128, 128, al, stream);
  scale_conv(p, x1, x2, 96, 192, 128, 128, 64, 64, al, stream);       // down
  for (int l = 0; l < 4; ++l) tlayer(p, x2, 192, 64, 64, al, stream);

  // ---- middle (tal x4 + up) ----
  for (int l = 0; l < 4; ++l) talayer(p, x2, 192, 64, 64, al, stream);
  scale_conv(p, x2, d1, 192, 96, 64, 64, 128, 128, al, stream);       // up

  // ---- decoders interleave with fuses (param order: decoders then fuses) ---
  // dec(ix=1): 2*tal(11) + up(2) = 24 tensors; dec(ix=0): 22;
  // fuse0(nf=48): 10; fuse1(nf=96): 10
  P pdec1 {d_in, p.i};
  P pdec0 {d_in, p.i + 24};
  P pfuse0{d_in, p.i + 46};
  P pfuse1{d_in, p.i + 56};
  p.i += 66;                                  // refine starts here

  fuse(pfuse1, d1, x1, f1, 96, 128, 128, al, stream);
  for (int l = 0; l < 2; ++l) talayer(pdec1, f1, 96, 128, 128, al, stream);
  scale_conv(pdec1, f1, d0, 96, 48, 128, 128, 256, 256, al, stream);  // up

  fuse(pfuse0, d0, x0, f0, 48, 256, 256, al, stream);
  for (int l = 0; l < 2; ++l) talayer(pdec0, f0, 48, 256, 256, al, stream);

  // ---- refine ----
  for (int l = 0; l < 2; ++l) talayer(p, f0, 48, 256, 256, al, stream);

  // ---- output: 3x3 conv 48 -> 3, + img residual ----
  { const float* w = p.n(); const float* b = p.n();
    k_conv3x3<<<cdiv_ll((long long)3 * 65536, 256), 256, 0, stream>>>(
        f0, w, b, (float*)d_out, 48, 3, 256, 256, img); }
}
